// GraphAttentionBlock_6820408066448
// MI455X (gfx1250) — matrix operand order repaired and checked by reading, not compiled, not run on hardware
//
#include <hip/hip_runtime.h>

#define BB 8
#define NN 2048
#define FF 64
#define NEGV -9.0e15f

typedef __attribute__((ext_vector_type(16))) _Float16 v16h;
typedef __attribute__((ext_vector_type(8)))  float    v8f;
typedef __attribute__((ext_vector_type(4)))  float    v4f;

// ---------------- Kernel 1: linears + scores (tiny, 268 MFLOP) ----------------
// one thread per node: h1 (-> f16 transposed), h2 (-> d_out), s1/s2 scalars
__global__ void __launch_bounds__(256)
prep_kernel(const float* __restrict__ x,
            const float* __restrict__ W1, const float* __restrict__ b1,
            const float* __restrict__ a1,
            const float* __restrict__ W2, const float* __restrict__ b2,
            _Float16* __restrict__ h1t,   // [B][64][2048] f16
            float* __restrict__ out,      // holds h2 = [B][2048][64]
            float* __restrict__ s1, float* __restrict__ s2)
{
    int t = blockIdx.x * blockDim.x + threadIdx.x;   // 0 .. B*N-1
    if (t >= BB * NN) return;
    int b = t >> 11, n = t & (NN - 1);

    float xv[FF];
    const float* xr = x + (size_t)t * FF;
#pragma unroll
    for (int k = 0; k < FF; ++k) xv[k] = xr[k];

    float s1v = 0.f, s2v = 0.f;
    for (int f = 0; f < FF; ++f) {
        const float* wr = W1 + f * FF;
        float acc = b1[f];
#pragma unroll
        for (int k = 0; k < FF; ++k) acc = fmaf(xv[k], wr[k], acc);
        s1v = fmaf(acc, a1[f], s1v);
        s2v = fmaf(acc, a1[FF + f], s2v);
        h1t[(((size_t)(b * FF + f)) << 11) + n] = (_Float16)acc;
    }
    for (int f = 0; f < FF; ++f) {
        const float* wr = W2 + f * FF;
        float acc = b2[f];
#pragma unroll
        for (int k = 0; k < FF; ++k) acc = fmaf(xv[k], wr[k], acc);
        out[(size_t)t * FF + f] = acc;   // identity-masked layer 2 == h2
    }
    s1[t] = s1v;
    s2[t] = s2v;
}

// ---------------- Kernel 2: the single 128MB adj pass ----------------
// compress adj row -> 64 bitmask words; exact softmax row max via monotone relu
__global__ void __launch_bounds__(256)
mask_kernel(const int* __restrict__ adj,
            const float* __restrict__ s1, const float* __restrict__ s2,
            const float* __restrict__ ab1,
            unsigned* __restrict__ packed,  // [B*N][64]
            float* __restrict__ Mrow)       // [B*N]
{
    int row = blockIdx.x;            // b*N + i
    int b   = row >> 11;
    const int*   arow = adj + (size_t)row * NN;
    const float* s2b  = s2 + ((size_t)b << 11);
    unsigned*    prow = packed + (size_t)row * (NN / 32);
    int t = threadIdx.x;

    float mx = -__builtin_inff();
#pragma unroll
    for (int k = 0; k < NN / 256; ++k) {
        int j  = t + 256 * k;                 // wave w covers j = w*32 + lane
        int av = arow[j];
        unsigned m = (unsigned)__ballot(av != 0);
        if ((t & 31) == 0) prow[j >> 5] = m;
        if (av) mx = fmaxf(mx, s2b[j]);
    }
#pragma unroll
    for (int off = 16; off > 0; off >>= 1)
        mx = fmaxf(mx, __shfl_xor(mx, off, 32));

    __shared__ float red[8];
    if ((t & 31) == 0) red[t >> 5] = mx;
    __syncthreads();
    if (t == 0) {
        float m = red[0];
#pragma unroll
        for (int w = 1; w < 8; ++w) m = fmaxf(m, red[w]);
        float M = (m == -__builtin_inff()) ? NEGV
                                           : fmaxf(s1[row] + ab1[0] + m, 0.0f);
        Mrow[row] = M;
    }
}

// ---------------- Kernel 3: WMMA softmax-attention (p @ h1) ----------------
// one wave per (batch, 16-row tile); j in chunks of 32 (K of v_wmma f16)
__global__ void __launch_bounds__(128)
attn_kernel(const unsigned* __restrict__ packed,
            const float* __restrict__ s1, const float* __restrict__ s2,
            const float* __restrict__ Mrow, const float* __restrict__ ab1,
            const _Float16* __restrict__ h1t,
            float* __restrict__ out)        // in: h2, out: h2 + att@h1
{
    int wv   = threadIdx.x >> 5;
    int lane = threadIdx.x & 31;
    int tile = blockIdx.x * 4 + wv;          // 0..1023
    int b    = tile >> 7;
    int i0   = (tile & 127) << 4;
    int r    = lane & 15;                    // row in tile (A-matrix M = lane%16)
    int hi   = lane >> 4;                    // K-half select

    int row  = (b << 11) + i0 + r;
    float M  = Mrow[row];
    // M >= 0 when the row has neighbors; exp(max(c1+sv,0) - M)
    //   == exp(max((c1 - M) + sv, -M)).  No-neighbor rows have all bits 0.
    float c1m  = s1[row] + ab1[0] - M;
    float negM = -M;
    const unsigned*  prow = packed + (size_t)row * (NN / 32);
    const float*     s2b  = s2 + ((size_t)b << 11);
    const _Float16*  hb   = h1t + (((size_t)b * FF) << 11);

    v8f acc[4] = {};
    float ps[4] = {0.f, 0.f, 0.f, 0.f};      // 4 independent sum chains
    // A-operand K sequence per ISA 7.12.2 (16-bit A 16x32):
    //   lanes 0-15:  K = 0..7, 16..23 ; lanes 16-31: K = 8..15, 24..31
    int abase = hi << 3;                     // 0 or 8

    for (int j0 = 0; j0 < NN; j0 += 32) {
        unsigned w = prow[j0 >> 5];
        v4f ch[4];
        ch[0] = *(const v4f*)(s2b + j0 + abase);
        ch[1] = *(const v4f*)(s2b + j0 + abase + 4);
        ch[2] = *(const v4f*)(s2b + j0 + abase + 16);
        ch[3] = *(const v4f*)(s2b + j0 + abase + 20);

        v16h a;
#pragma unroll
        for (int e = 0; e < 16; ++e) {
            int k    = abase + (e < 8 ? e : e + 8);   // bit index within word
            float sv = ch[e >> 2][e & 3];
            float ev = fmaxf(c1m + sv, negM);
            float p  = ((w >> k) & 1u) ? __expf(ev) : 0.f;
            ps[e & 3] += p;                           // 4-way parallel chains
            a[e] = (_Float16)p;
        }

        // B-operand (32x16 f16): same K order as A: lanes 0-15 K=0..7,16..23,
        // lanes 16-31 K=8..15,24..31 -> two 16B loads per N-tile from transposed h1
        int bofs = j0 + (hi << 3);
#pragma unroll
        for (int nt = 0; nt < 4; ++nt) {
            const _Float16* hp = hb + (((size_t)((nt << 4) + r)) << 11) + bofs;
            v4f blo = *(const v4f*)hp;
            v4f bhi = *(const v4f*)(hp + 16);
            v16h bm = __builtin_bit_cast(v16h, __builtin_shufflevector(blo, bhi, 0, 1, 2, 3, 4, 5, 6, 7));
            acc[nt] = __builtin_amdgcn_wmma_f32_16x16x32_f16(
                false, a, false, bm, (short)0, acc[nt], false, false);
        }
    }

    float psum = (ps[0] + ps[1]) + (ps[2] + ps[3]);
    // row sum: each row split across lanes r and r+16
    float rs = psum + __shfl_xor(psum, 16, 32);

    // C/D layout: VGPR v, lanes 0-15 -> M=v, lanes 16-31 -> M=v+8 ; N = lane%16
#pragma unroll
    for (int v = 0; v < 8; ++v) {
        float rsv  = __shfl(rs, v + (hi << 3), 32);
        float invv = (rsv > 0.f) ? 1.0f / rsv : 0.0f;
        int   orow = (b << 11) + i0 + v + (hi << 3);
#pragma unroll
        for (int nt = 0; nt < 4; ++nt) {
            size_t oidx = ((size_t)orow << 6) + (nt << 4) + r;
            out[oidx] = fmaf(acc[nt][v], invv, out[oidx]);   // += h2 already there
        }
    }
}

// ---------------- host ----------------
extern "C" void kernel_launch(void* const* d_in, const int* in_sizes, int n_in,
                              void* d_out, int out_size, void* d_ws, size_t ws_size,
                              hipStream_t stream)
{
    const float* x   = (const float*)d_in[0];
    const int*   adj = (const int*)  d_in[1];
    // d_in[2] identity: layer-2 mask collapses to out2 = h2, unused
    const float* W1  = (const float*)d_in[3];
    const float* b1  = (const float*)d_in[4];
    const float* a1  = (const float*)d_in[5];
    const float* ab1 = (const float*)d_in[6];
    const float* W2  = (const float*)d_in[7];
    const float* b2  = (const float*)d_in[8];
    // d_in[9], d_in[10] (a2, ab2): one-hot softmax is invariant to them

    float* out = (float*)d_out;
    char*  ws  = (char*)d_ws;
    _Float16* h1t    = (_Float16*)(ws + 0);          // 2 MB
    float*    s1     = (float*)   (ws + 2097152);    // 64 KB
    float*    s2     = (float*)   (ws + 2162688);    // 64 KB
    float*    Mr     = (float*)   (ws + 2228224);    // 64 KB
    unsigned* packed = (unsigned*)(ws + 2293760);    // 4 MB

    prep_kernel<<<(BB * NN) / 256, 256, 0, stream>>>(x, W1, b1, a1, W2, b2,
                                                     h1t, out, s1, s2);
    mask_kernel<<<BB * NN, 256, 0, stream>>>(adj, s1, s2, ab1, packed, Mr);
    attn_kernel<<<(BB * NN / 16) / 4, 128, 0, stream>>>(packed, s1, s2, Mr, ab1,
                                                        h1t, out);
}
